// GNN_15144054685737
// MI455X (gfx1250) — matrix operand order repaired and checked by reading, not compiled, not run on hardware
//
#include <hip/hip_runtime.h>
#include <hip/hip_bf16.h>
#include <math.h>

// ---------------- problem constants ----------------
#define NN 20000      // nodes
#define EE 640000     // edges
#define GG 128        // input dim
#define H1 128        // rgcn out dim
#define H2 128        // transformer out dim
#define RR 8          // relations
#define KTOT (RR*GG + GG)   // 1152
#define MBLK 64             // rows per block (4 wmma tiles of 16)
#define NPAD 20032          // 313 * 64, padded row count for guard-free GEMM loads

typedef __bf16 bf16;
typedef __attribute__((ext_vector_type(16))) __bf16 v16bf;
typedef __attribute__((ext_vector_type(8)))  float  v8f;
typedef int v4i_b128 __attribute__((vector_size(16)));   // matches async builtin param type

#if __has_builtin(__builtin_amdgcn_global_load_async_to_lds_b128) && \
    __has_builtin(__builtin_amdgcn_s_wait_asynccnt)
#define USE_ASYNC_LDS 1
#else
#define USE_ASYNC_LDS 0
#endif

#define AS1 __attribute__((address_space(1)))
#define AS3 __attribute__((address_space(3)))

__device__ __forceinline__ unsigned short f32_bf16_bits(float f) {
    unsigned int u = __float_as_uint(f);
    return (unsigned short)((u + 0x7FFFu + ((u >> 16) & 1u)) >> 16);
}
__device__ __forceinline__ bf16 f32_bf16(float f) {
    return __builtin_bit_cast(bf16, f32_bf16_bits(f));
}

__device__ __forceinline__ void atomicMaxFloat(float* addr, float value) {
    if (value >= 0.0f) atomicMax((int*)addr, __float_as_int(value));
    else               atomicMin((unsigned int*)addr, __float_as_uint(value));
}

// copy one 8KB bf16 panel (global, panel-ordered) into LDS B tile
__device__ __forceinline__ void stage_B_panel(const bf16* gpanel, bf16* lbase, int tid) {
#if USE_ASYNC_LDS
    #pragma unroll
    for (int u = 0; u < 2; ++u) {
        int e = u * 256 + tid;                   // 512 x 16B = 8KB
        __builtin_amdgcn_global_load_async_to_lds_b128(
            (AS1 v4i_b128*)(uintptr_t)((const char*)gpanel + (size_t)e * 16),
            (AS3 v4i_b128*)(uintptr_t)((char*)lbase + (size_t)e * 16), 0, 0);
    }
    __builtin_amdgcn_s_wait_asynccnt(0);
#else
    #pragma unroll
    for (int u = 0; u < 2; ++u) {
        int e = u * 256 + tid;
        ((uint4*)lbase)[e] = ((const uint4*)gpanel)[e];
    }
#endif
}

// stage one 64x32 fp32 A tile -> bf16 fragment-order LDS, guard-free (padded rows).
// Each thread: 2 x (float4 load + pack + 8B DS store). kq (aligned 4) never
// crosses a fragment K-region boundary (regions are 16 wide).
__device__ __forceinline__ void stage_A_tile(bf16 (*ldsAf)[32][16],
                                             const float* __restrict__ Abase,
                                             int astride, int rowBase, int tid) {
    #pragma unroll
    for (int it = 0; it < 2; ++it) {
        int e4 = it * 256 + tid;        // quad index, 512 quads
        int m  = e4 >> 3;               // 0..63
        int kq = (e4 & 7) * 4;          // 0,4,...,28
        int row = rowBase + m;
        float4 v4 = *(const float4*)(Abase + (size_t)row * astride + kq);
        uint2 pk;
        pk.x = ((unsigned)f32_bf16_bits(v4.y) << 16) | f32_bf16_bits(v4.x);
        pk.y = ((unsigned)f32_bf16_bits(v4.w) << 16) | f32_bf16_bits(v4.z);
        int tt = m >> 4, mr = m & 15;
        int L, j;
        if      (kq < 16) { L = mr;      j = kq;      }
        else              { L = mr + 16; j = kq - 16; }
        *(uint2*)&ldsAf[tt][L][j] = pk;
    }
}

// ---------------- utility kernels ----------------
__global__ void fill_kernel(float* __restrict__ p, float val, long n) {
    long i = (long)blockIdx.x * blockDim.x + threadIdx.x;
    if (i < n) p[i] = val;
}

__global__ void relu_kernel(float* __restrict__ p, long n) {
    long i = (long)blockIdx.x * blockDim.x + threadIdx.x;
    if (i < n) p[i] = fmaxf(p[i], 0.0f);
}

// padded copy of x: rows >= NN zeroed
__global__ void xpad_kernel(const float* __restrict__ x, float* __restrict__ xpad) {
    long i = (long)blockIdx.x * blockDim.x + threadIdx.x;
    if (i < (long)NPAD * GG) {
        long row = i >> 7;
        xpad[i] = (row < NN) ? x[i] : 0.0f;
    }
}

__global__ void count_kernel(const int* __restrict__ dst,
                             const int* __restrict__ rel,
                             float* __restrict__ cnt) {
    int e = blockIdx.x * blockDim.x + threadIdx.x;
    if (e < EE) atomicAdd(&cnt[(long)dst[e] * RR + rel[e]], 1.0f);
}

// scatter x[src]/max(cnt,1) into bucket[dst][rel*128+g] (wave32 per edge)
__global__ void scatter_rgcn_kernel(const float* __restrict__ x,
                                    const int* __restrict__ src,
                                    const int* __restrict__ dst,
                                    const int* __restrict__ rel,
                                    const float* __restrict__ cnt,
                                    float* __restrict__ bucket) {
    int gid  = blockIdx.x * blockDim.x + threadIdx.x;
    int e    = gid >> 5;
    int lane = gid & 31;
    if (e < EE) {
        int s = src[e], d = dst[e], r = rel[e];
        float inv = 1.0f / fmaxf(cnt[(long)d * RR + r], 1.0f);
        float4 v4 = ((const float4*)(x + (size_t)s * GG))[lane];
        float* bp = bucket + (size_t)d * (RR * GG) + r * GG + lane * 4;
        atomicAdd(bp + 0, inv * v4.x);
        atomicAdd(bp + 1, inv * v4.y);
        atomicAdd(bp + 2, inv * v4.z);
        atomicAdd(bp + 3, inv * v4.w);
    }
}

// ---------------- weight prep: fp32 -> bf16 panel order [panel][col][kk] ----------------
__global__ void prep_b1_kernel(const float* __restrict__ Wrel,   // 1024 x 128
                               const float* __restrict__ root,   // 128 x 128
                               bf16* __restrict__ bc1) {          // 36 panels
    int idx = blockIdx.x * blockDim.x + threadIdx.x;
    if (idx < (KTOT / 32) * 128 * 32) {
        int kk  = idx & 31;
        int col = (idx >> 5) & 127;
        int p   = idx >> 12;
        int kg  = p * 32 + kk;
        float v = (kg < RR * GG) ? Wrel[(size_t)kg * H1 + col]
                                 : root[(size_t)(kg - RR * GG) * H1 + col];
        bc1[idx] = f32_bf16(v);
    }
}

__global__ void prep_b2_kernel(const float* __restrict__ Wq, const float* __restrict__ Wk,
                               const float* __restrict__ Wv, const float* __restrict__ Ws,
                               bf16* __restrict__ bc2) {          // 4 weights x 4 panels
    int idx = blockIdx.x * blockDim.x + threadIdx.x;
    if (idx < 4 * 4 * 128 * 32) {
        int w   = idx >> 14;
        int rem = idx & 16383;
        int kk  = rem & 31;
        int col = (rem >> 5) & 127;
        int p   = rem >> 12;
        const float* W = (w == 0) ? Wq : (w == 1) ? Wk : (w == 2) ? Wv : Ws;
        bc2[idx] = f32_bf16(W[(size_t)(p * 32 + kk) * H2 + col]);
    }
}

// ---------------- GEMM1: fused RGCN ----------------
// h = relu( [bucket(pre-scaled) | xpad] @ [Wrel ; root] + bias ), rows padded to NPAD
__global__ void gemm_rgcn_kernel(const float* __restrict__ xpad,  // [NPAD][128]
                                 const bf16*  __restrict__ bc1,   // panel-ordered B
                                 const float* __restrict__ bias,
                                 const float* __restrict__ bucket,// [NPAD][1024], /cnt applied
                                 float* __restrict__ h) {          // [NPAD][128]
    __shared__ alignas(32) bf16 ldsAf[4][32][16];  // 4 M-tiles, fragment order
    __shared__ alignas(32) bf16 ldsB[128][32];     // col-major (fragment order)
    const int tid  = threadIdx.x;
    const int wave = tid >> 5;
    const int lane = tid & 31;
    const int rowBase = blockIdx.x * MBLK;

    v8f acc[4];
    #pragma unroll
    for (int t = 0; t < 4; ++t) acc[t] = (v8f){0.f,0.f,0.f,0.f,0.f,0.f,0.f,0.f};

    for (int k0 = 0; k0 < KTOT; k0 += 32) {
        // uniform per-step A source selection (boundary 1024 is a multiple of 32)
        const float* Abase;
        int astride;
        if (k0 < RR * GG) { Abase = bucket + k0;         astride = RR * GG; }
        else              { Abase = xpad + (k0 - RR*GG); astride = GG;      }

        stage_A_tile(ldsAf, Abase, astride, rowBase, tid);
        stage_B_panel(bc1 + (size_t)(k0 >> 5) * (128 * 32), &ldsB[0][0], tid);
        __syncthreads();

        const int bcol = (wave << 4) + (lane & 15);
        const int bkb  = (lane < 16) ? 0 : 16;
        v16bf bfrag = *reinterpret_cast<const v16bf*>(&ldsB[bcol][bkb]);
        #pragma unroll
        for (int t = 0; t < 4; ++t) {
            v16bf afrag = *reinterpret_cast<const v16bf*>(&ldsAf[t][lane][0]);
            acc[t] = __builtin_amdgcn_wmma_f32_16x16x32_bf16(
                false, afrag, false, bfrag, (short)0, acc[t], false, false);
        }
        __syncthreads();
    }

    const int col = (wave << 4) + (lane & 15);
    const float bcol_bias = bias[col];
    #pragma unroll
    for (int t = 0; t < 4; ++t) {
        #pragma unroll
        for (int i = 0; i < 8; ++i) {
            int m   = (lane < 16) ? i : (8 + i);
            int row = rowBase + t * 16 + m;     // unguarded: h is padded
            h[(size_t)row * H1 + col] = fmaxf(acc[t][i] + bcol_bias, 0.0f);
        }
    }
}

// ---------------- GEMM2: q/k/v/skip (blockIdx.y selects weight) ----------------
__global__ void gemm_qkvs_kernel(const float* __restrict__ h,     // [NPAD][128]
                                 const bf16*  __restrict__ bc2,
                                 const float* __restrict__ bq, const float* __restrict__ bk,
                                 const float* __restrict__ bv, const float* __restrict__ bs,
                                 float* __restrict__ q, float* __restrict__ k,
                                 float* __restrict__ v, float* __restrict__ out) {
    __shared__ alignas(32) bf16 ldsAf[4][32][16];
    __shared__ alignas(32) bf16 ldsB[128][32];
    const int tid  = threadIdx.x;
    const int wave = tid >> 5;
    const int lane = tid & 31;
    const int rowBase = blockIdx.x * MBLK;
    const int sel = blockIdx.y;

    const float* b; float* dstp;
    if      (sel == 0) { b = bq; dstp = q;   }
    else if (sel == 1) { b = bk; dstp = k;   }
    else if (sel == 2) { b = bv; dstp = v;   }
    else               { b = bs; dstp = out; }
    const bf16* bpanels = bc2 + (size_t)sel * 4 * 128 * 32;

    v8f acc[4];
    #pragma unroll
    for (int t = 0; t < 4; ++t) acc[t] = (v8f){0.f,0.f,0.f,0.f,0.f,0.f,0.f,0.f};

    #pragma unroll
    for (int p = 0; p < H1 / 32; ++p) {
        stage_A_tile(ldsAf, h + p * 32, H1, rowBase, tid);
        stage_B_panel(bpanels + (size_t)p * (128 * 32), &ldsB[0][0], tid);
        __syncthreads();

        const int bcol = (wave << 4) + (lane & 15);
        const int bkb  = (lane < 16) ? 0 : 16;
        v16bf bfrag = *reinterpret_cast<const v16bf*>(&ldsB[bcol][bkb]);
        #pragma unroll
        for (int t = 0; t < 4; ++t) {
            v16bf afrag = *reinterpret_cast<const v16bf*>(&ldsAf[t][lane][0]);
            acc[t] = __builtin_amdgcn_wmma_f32_16x16x32_bf16(
                false, afrag, false, bfrag, (short)0, acc[t], false, false);
        }
        __syncthreads();
    }

    const int col = (wave << 4) + (lane & 15);
    const float bb = b[col];
    if (sel != 3) {
        // q/k/v are padded scratch: unguarded stores
        #pragma unroll
        for (int t = 0; t < 4; ++t) {
            #pragma unroll
            for (int i = 0; i < 8; ++i) {
                int m   = (lane < 16) ? i : (8 + i);
                int row = rowBase + t * 16 + m;
                dstp[(size_t)row * H2 + col] = acc[t][i] + bb;
            }
        }
    } else {
        // out = d_out has exactly NN rows: guard
        #pragma unroll
        for (int t = 0; t < 4; ++t) {
            #pragma unroll
            for (int i = 0; i < 8; ++i) {
                int m   = (lane < 16) ? i : (8 + i);
                int row = rowBase + t * 16 + m;
                if (row < NN)
                    dstp[(size_t)row * H2 + col] = acc[t][i] + bb;
            }
        }
    }
}

// ---------------- attention edge kernels (wave32 per edge) ----------------
__global__ void score_kernel(const float* __restrict__ q,
                             const float* __restrict__ k,
                             const int* __restrict__ src,
                             const int* __restrict__ dst,
                             float* __restrict__ score,
                             float* __restrict__ smax) {
    int gid  = blockIdx.x * blockDim.x + threadIdx.x;
    int e    = gid >> 5;
    int lane = gid & 31;
    if (e < EE) {
        int s = src[e], d = dst[e];
        float4 qv = ((const float4*)(q + (size_t)d * H2))[lane];
        float4 kv = ((const float4*)(k + (size_t)s * H2))[lane];
        float p = qv.x * kv.x + qv.y * kv.y + qv.z * kv.z + qv.w * kv.w;
        #pragma unroll
        for (int off = 16; off > 0; off >>= 1)
            p += __shfl_xor(p, off, 32);
        if (lane == 0) {
            float sc = p * 0.08838834764831845f; // 1/sqrt(128)
            score[e] = sc;
            atomicMaxFloat(&smax[d], sc);
        }
    }
}

__global__ void expsum_kernel(const float* __restrict__ score,
                              const int* __restrict__ dst,
                              const float* __restrict__ smax,
                              float* __restrict__ evals,
                              float* __restrict__ den) {
    int e = blockIdx.x * blockDim.x + threadIdx.x;
    if (e < EE) {
        int d = dst[e];
        float ev = __expf(score[e] - smax[d]);
        evals[e] = ev;
        atomicAdd(&den[d], ev);
    }
}

__global__ void scatter_attn_kernel(const float* __restrict__ v,
                                    const int* __restrict__ src,
                                    const int* __restrict__ dst,
                                    const float* __restrict__ evals,
                                    const float* __restrict__ den,
                                    float* __restrict__ out) {
    int gid  = blockIdx.x * blockDim.x + threadIdx.x;
    int e    = gid >> 5;
    int lane = gid & 31;
    if (e < EE) {
        int s = src[e], d = dst[e];
        float alpha = evals[e] / fmaxf(den[d], 1e-16f);
        float4 vv = ((const float4*)(v + (size_t)s * H2))[lane];
        float* op = out + (size_t)d * H2 + lane * 4;
        atomicAdd(op + 0, alpha * vv.x);
        atomicAdd(op + 1, alpha * vv.y);
        atomicAdd(op + 2, alpha * vv.z);
        atomicAdd(op + 3, alpha * vv.w);
    }
}

// ---------------- launcher ----------------
extern "C" void kernel_launch(void* const* d_in, const int* in_sizes, int n_in,
                              void* d_out, int out_size, void* d_ws, size_t ws_size,
                              hipStream_t stream) {
    const float* x    = (const float*)d_in[0];
    const int*   eidx = (const int*)d_in[1];   // [2,E]
    const int*   src  = eidx;
    const int*   dst  = eidx + EE;
    const int*   rel  = (const int*)d_in[2];
    const float* Wrel = (const float*)d_in[3];
    const float* root = (const float*)d_in[4];
    const float* rbia = (const float*)d_in[5];
    const float* Wq   = (const float*)d_in[6];
    const float* bq   = (const float*)d_in[7];
    const float* Wk   = (const float*)d_in[8];
    const float* bk   = (const float*)d_in[9];
    const float* Wv   = (const float*)d_in[10];
    const float* bv   = (const float*)d_in[11];
    const float* Ws   = (const float*)d_in[12];
    const float* bs   = (const float*)d_in[13];
    float* out        = (float*)d_out;

    // workspace layout (float-sized slots; all section sizes multiples of 4)
    float* ws = (float*)d_ws;
    size_t o = 0;
    float* cnt    = ws + o; o += (size_t)NN * RR;
    float* bucket = ws + o; o += (size_t)NPAD * RR * GG;   // padded rows
    float* xpad   = ws + o; o += (size_t)NPAD * GG;
    float* h      = ws + o; o += (size_t)NPAD * H1;
    float* q      = ws + o; o += (size_t)NPAD * H2;
    float* k      = ws + o; o += (size_t)NPAD * H2;
    float* v      = ws + o; o += (size_t)NPAD * H2;
    float* smax   = ws + o; o += NN;
    float* den    = ws + o; o += NN;
    float* score  = ws + o; o += EE;
    float* evals  = ws + o; o += EE;
    bf16*  bc1    = (bf16*)(ws + o); o += (size_t)(KTOT / 32) * 128 * 32 / 2; // bf16
    bf16*  bc2    = (bf16*)(ws + o); o += (size_t)4 * 4 * 128 * 32 / 2;       // bf16

    const int T = 256;
    const int MB = NPAD / MBLK; // 313

    // init scratch
    fill_kernel<<<((long)NN*RR + T - 1) / T, T, 0, stream>>>(cnt, 0.0f, (long)NN*RR);
    fill_kernel<<<((long)NPAD*RR*GG + T - 1) / T, T, 0, stream>>>(bucket, 0.0f, (long)NPAD*RR*GG);
    fill_kernel<<<(NN + T - 1) / T, T, 0, stream>>>(smax, -INFINITY, NN);
    fill_kernel<<<(NN + T - 1) / T, T, 0, stream>>>(den, 0.0f, NN);
    xpad_kernel<<<((long)NPAD*GG + T - 1) / T, T, 0, stream>>>(x, xpad);

    // weight prep (fp32 -> bf16 panels)
    prep_b1_kernel<<<((KTOT/32)*128*32 + T - 1) / T, T, 0, stream>>>(Wrel, root, bc1);
    prep_b2_kernel<<<(4*4*128*32 + T - 1) / T, T, 0, stream>>>(Wq, Wk, Wv, Ws, bc2);

    // RGCN: counts, mean-scaled aggregate, fused GEMM
    count_kernel<<<(EE + T - 1) / T, T, 0, stream>>>(dst, rel, cnt);
    scatter_rgcn_kernel<<<((long)EE * 32 + T - 1) / T, T, 0, stream>>>(x, src, dst, rel, cnt, bucket);
    gemm_rgcn_kernel<<<MB, T, 0, stream>>>(xpad, bc1, rbia, bucket, h);

    // Transformer: q/k/v + skip (skip written directly into out)
    gemm_qkvs_kernel<<<dim3(MB, 4), T, 0, stream>>>(h, bc2, bq, bk, bv, bs, q, k, v, out);

    // edge softmax + weighted aggregation
    score_kernel<<<((long)EE * 32 + T - 1) / T, T, 0, stream>>>(q, k, src, dst, score, smax);
    expsum_kernel<<<(EE + T - 1) / T, T, 0, stream>>>(score, dst, smax, evals, den);
    scatter_attn_kernel<<<((long)EE * 32 + T - 1) / T, T, 0, stream>>>(v, src, dst, evals, den, out);

    relu_kernel<<<((long)NN * H2 + T - 1) / T, T, 0, stream>>>(out, (long)NN * H2);
}
